// EMA_VQEmbedding_67216238182695
// MI455X (gfx1250) — matrix operand order repaired and checked by reading, not compiled, not run on hardware
//
#include <hip/hip_runtime.h>
#include <hip/hip_bf16.h>

#define K_CODES 1024
#define DIM     256
#define NROWS   65536            // 16 * 4096
#define DECAYC  0.99f
#define COMMITC 0.25f
#define EPSV    1e-5f
#define TILE_CODES 64
#define NCHUNKS (K_CODES / TILE_CODES)      // 16
#define BUF_USHORTS 32768                    // 64KB per double-buffer slot (hi+lo)
#define HALF_USHORTS 16384                   // 32KB: one bf16 slice [DIM][TILE_CODES]

typedef __attribute__((ext_vector_type(16))) __bf16 v16bf;
typedef __attribute__((ext_vector_type(8)))  float  v8f;
typedef __attribute__((ext_vector_type(4)))  unsigned int u32x4;
typedef __attribute__((ext_vector_type(8)))  int  i32x8;
typedef __attribute__((ext_vector_type(4)))  int  i32x4;

union AFrag { v16bf v; unsigned int u[8]; };
union BFrag { v16bf v; unsigned int u[8]; uint4 q[2]; };

__device__ __forceinline__ unsigned short f2bf(float f) {
    unsigned int u = __float_as_uint(f);
    unsigned int r = u + 0x7FFFu + ((u >> 16) & 1u);   // round-to-nearest-even
    return (unsigned short)(r >> 16);
}
__device__ __forceinline__ float bf2f(unsigned short h) {
    return __uint_as_float(((unsigned int)h) << 16);
}

// ---- Prep: split inputs into bf16 hi/lo -------------------------------------
__global__ void k_prep_x(const float* __restrict__ x,
                         unsigned short* __restrict__ xhi,
                         unsigned short* __restrict__ xlo, int n) {
    int i = blockIdx.x * blockDim.x + threadIdx.x;
    if (i < n) {
        float v = x[i];
        unsigned short h = f2bf(v);
        xhi[i] = h;
        xlo[i] = f2bf(v - bf2f(h));
    }
}

// ---- Prep: split + transpose codebook to [D][K] -----------------------------
__global__ void k_prep_w(const float* __restrict__ w,
                         unsigned short* __restrict__ wthi,
                         unsigned short* __restrict__ wtlo) {
    int i = blockIdx.x * blockDim.x + threadIdx.x;   // over K*D
    if (i < K_CODES * DIM) {
        int k = i / DIM, d = i % DIM;
        float v = w[i];
        unsigned short h = f2bf(v);
        wthi[d * K_CODES + k] = h;
        wtlo[d * K_CODES + k] = f2bf(v - bf2f(h));
    }
}

// ---- Prep: |e_k|^2 in exact fp32 -------------------------------------------
__global__ void k_wsq(const float* __restrict__ w, float* __restrict__ wsq) {
    __shared__ float red[8];
    int k = blockIdx.x, d = threadIdx.x;
    float v = w[k * DIM + d];
    float s = v * v;
    for (int m = 16; m >= 1; m >>= 1) s += __shfl_xor(s, m, 32);
    if ((d & 31) == 0) red[d >> 5] = s;
    __syncthreads();
    if (d < 8) {
        float t = red[d];
        for (int m = 4; m >= 1; m >>= 1) t += __shfl_xor(t, m, 8);
        if (d == 0) wsq[k] = t;
    }
}

// ---- TDM: issue one 2D tile load (codebook slice [DIM][TILE_CODES] bf16) ----
// D# per ISA 8.3/8.4: count=1, type=2; data_size=1 (2B);
// tensor_dim0=K (1024), tensor_dim1=DIM (256), tensor_dim0_stride=K;
// tile_dim0=TILE_CODES (64), tile_dim1=DIM (256). TDM packs rows contiguously
// into LDS -> exactly our [DIM][TILE_CODES] staging layout.
__device__ __forceinline__ void tdm_issue(const unsigned short* gsrc,
                                          unsigned int ldsByteAddr) {
    unsigned long long ga = (unsigned long long)gsrc;
    u32x4 g0 = { 1u,                                   // count=1, user desc
                 ldsByteAddr,                          // lds_addr
                 (unsigned int)ga,                     // global_addr[31:0]
                 (unsigned int)((ga >> 32) & 0x1FFFFFFull) | (2u << 30) }; // addr[56:32] | type=2
    i32x8 g1 = { (int)(1u << 16),                      // data_size=1 (2 bytes)
                 (int)(1024u << 16),                   // tensor_dim0[15:0]=1024
                 (int)(256u << 16),                    // tensor_dim0[31:16]=0 | tensor_dim1[15:0]=256
                 (int)(64u << 16),                     // tensor_dim1[31:16]=0 | tile_dim0=64
                 256,                                  // tile_dim1=256, tile_dim2=0
                 1024,                                 // tensor_dim0_stride[31:0]=1024
                 0, 0 };
    i32x4 z4 = { 0, 0, 0, 0 };
    i32x8 z8 = { 0, 0, 0, 0, 0, 0, 0, 0 };
    // amdgpu-toolchain (clang-23) 6-arg form: (g0, g1, g2, g3, g4, cpol)
    __builtin_amdgcn_tensor_load_to_lds(g0, g1, z4, z4, z8, 0);
}

// ---- Main: TDM-fed, double-buffered WMMA distance GEMM + fused argmin -------
// Block = 256 threads (8 waves), each wave owns 16 rows (A frags in VGPRs).
// Wave 0 drives the Tensor Data Mover; all waves consume via LDS.
__global__ void __launch_bounds__(256) k_argmin(
    const unsigned short* __restrict__ xhi, const unsigned short* __restrict__ xlo,
    const unsigned short* __restrict__ wthi, const unsigned short* __restrict__ wtlo,
    const float* __restrict__ wsq, int* __restrict__ idx_out)
{
    extern __shared__ unsigned short lds[];           // 2*64KB tiles + 4KB wsq
    float* lwsq = (float*)(lds + 2 * BUF_USHORTS);    // byte offset 131072

    const int lane = threadIdx.x & 31;
    const int wave = threadIdx.x >> 5;
    const int m    = lane & 15;       // A: row-within-tile; C: N column
    const int half = lane >> 4;
    const int rowBase = blockIdx.x * 128 + wave * 16;

    // Stage |e|^2 into LDS once (removes LOADcnt stalls from the hot loop).
    for (int i = threadIdx.x; i < K_CODES; i += 256) lwsq[i] = wsq[i];

    // Preload A fragments: 8 K-chunks of 32, hi and lo (ISA 16-bit A layout:
    // lane -> M = lane%16; VGPR pairs hold K = (i&8)*2 + half*8 + (i&7)).
    AFrag ahi[8], alo[8];
    const unsigned short* xh = xhi + (size_t)(rowBase + m) * DIM;
    const unsigned short* xl = xlo + (size_t)(rowBase + m) * DIM;
    #pragma unroll
    for (int c = 0; c < 8; ++c) {
        int dbase = c * 32;
        #pragma unroll
        for (int i = 0; i < 16; i += 2) {
            int k = ((i & 8) << 1) + half * 8 + (i & 7);   // even -> dword load
            ahi[c].u[i >> 1] = *(const unsigned int*)(xh + dbase + k);
            alo[c].u[i >> 1] = *(const unsigned int*)(xl + dbase + k);
        }
    }

    // TDM prologue: chunk 0 -> buffer 0 (wave0 only; EXEC-independent issue).
    const unsigned int ldsBase = (unsigned int)(unsigned long long)(void*)lds;
    if (wave == 0) {
        tdm_issue(wthi, ldsBase);
        tdm_issue(wtlo, ldsBase + 2u * HALF_USHORTS);
    }

    float runv[8];
    int   runi[8];
    #pragma unroll
    for (int r = 0; r < 8; ++r) { runv[r] = 3.4e38f; runi[r] = 0; }

    for (int ic = 0; ic < NCHUNKS; ++ic) {
        const int kc = ic * TILE_CODES;
        if (wave == 0) {
            if (ic + 1 < NCHUNKS) {
                // Prefetch next chunk, then wait until only those 2 ops remain
                // in flight (=> current chunk's TDM writes have landed).
                unsigned int nb = ldsBase + (unsigned)((ic + 1) & 1) * (2u * BUF_USHORTS);
                tdm_issue(wthi + (kc + TILE_CODES), nb);
                tdm_issue(wtlo + (kc + TILE_CODES), nb + 2u * HALF_USHORTS);
                __builtin_amdgcn_s_wait_tensorcnt((short)2);
            } else {
                __builtin_amdgcn_s_wait_tensorcnt((short)0);
            }
        }
        __syncthreads();                               // chunk ic visible to all
        const unsigned short* bufp = lds + (ic & 1) * BUF_USHORTS;

        #pragma unroll
        for (int nt = 0; nt < 4; ++nt) {
            int code0  = kc + nt * 16;
            int mycode = code0 + m;
            float bias = -0.5f * lwsq[mycode];         // ds_load, no vmem stall
            // Three independent accumulation chains for the XDL pipe.
            v8f acc0, acc1, acc2;
            #pragma unroll
            for (int r = 0; r < 8; ++r) { acc0[r] = bias; acc1[r] = 0.f; acc2[r] = 0.f; }

            #pragma unroll
            for (int c = 0; c < 8; ++c) {
                // B layout: lane -> N (code m), elements -> K (d) in same order as A
                BFrag bh, bl;
                #pragma unroll
                for (int j = 0; j < 8; ++j) {
                    int kk = ((j & 4) << 2) + half * 8 + 2 * (j & 3);
                    const unsigned short* ph = bufp + (c * 32 + kk) * TILE_CODES + nt * 16 + m;
                    const unsigned short* pl = ph + HALF_USHORTS;
                    bh.u[j] = (unsigned int)ph[0] | ((unsigned int)ph[TILE_CODES] << 16);
                    bl.u[j] = (unsigned int)pl[0] | ((unsigned int)pl[TILE_CODES] << 16);
                }
                acc0 = __builtin_amdgcn_wmma_f32_16x16x32_bf16(false, ahi[c].v, false, bh.v, (short)0, acc0, false, false);
                acc1 = __builtin_amdgcn_wmma_f32_16x16x32_bf16(false, ahi[c].v, false, bl.v, (short)0, acc1, false, false);
                acc2 = __builtin_amdgcn_wmma_f32_16x16x32_bf16(false, alo[c].v, false, bh.v, (short)0, acc2, false, false);
            }
            #pragma unroll
            for (int r = 0; r < 8; ++r) {
                float dv = -2.0f * (acc0[r] + (acc1[r] + acc2[r]));
                if (dv < runv[r] || (dv == runv[r] && mycode < runi[r])) {
                    runv[r] = dv; runi[r] = mycode;
                }
            }
        }
        __syncthreads();    // all reads of this buffer done before TDM reuses it
    }

    // Cross-lane argmin within each 16-lane half (C layout: VGPR r -> M = r + half*8).
    #pragma unroll
    for (int r = 0; r < 8; ++r) {
        float v = runv[r];
        int  ix = runi[r];
        #pragma unroll
        for (int msk = 8; msk >= 1; msk >>= 1) {
            float ov = __shfl_xor(v, msk, 16);
            int   oi = __shfl_xor(ix, msk, 16);
            if (ov < v || (ov == v && oi < ix)) { v = ov; ix = oi; }
        }
        if (m == 0) idx_out[rowBase + r + half * 8] = ix;
    }
}

// ---- Gather + SSE partials + segment sums -----------------------------------
__global__ void k_gather(const float* __restrict__ x, const float* __restrict__ w,
                         const int* __restrict__ idx,
                         float* __restrict__ outq, float* __restrict__ outidxf,
                         float* __restrict__ counts, float* __restrict__ embsum,
                         float* __restrict__ ssepart)
{
    __shared__ float red[8];
    int row = blockIdx.x;
    int d = threadIdx.x;
    int k = idx[row];
    float xv = x[(size_t)row * DIM + d];
    float qv = w[(size_t)k * DIM + d];
    outq[(size_t)row * DIM + d] = qv;          // quantized_st == quantized numerically
    float diff = qv - xv;
    float s = diff * diff;
    atomicAdd(&embsum[(size_t)k * DIM + d], xv);
    for (int m2 = 16; m2 >= 1; m2 >>= 1) s += __shfl_xor(s, m2, 32);
    if ((d & 31) == 0) red[d >> 5] = s;
    __syncthreads();
    if (d < 8) {
        float t = red[d];
        for (int m2 = 4; m2 >= 1; m2 >>= 1) t += __shfl_xor(t, m2, 8);
        if (d == 0) {
            ssepart[row] = t;
            atomicAdd(&counts[k], 1.0f);       // exact: integer-valued float adds
            outidxf[row] = (float)k;
        }
    }
}

// ---- Finalize: EMA update, normalization, loss ------------------------------
__global__ void __launch_bounds__(1024) k_final(
    const float* __restrict__ counts, const float* __restrict__ embsum,
    const float* __restrict__ ema_cs, const float* __restrict__ ema_avg,
    const float* __restrict__ ssepart,
    float* __restrict__ out_loss, float* __restrict__ out_w,
    float* __restrict__ out_cs, float* __restrict__ out_avg)
{
    __shared__ float lcs[1024];
    __shared__ float lsse[32];
    int t = threadIdx.x;
    float ncs = DECAYC * ema_cs[t] + (1.0f - DECAYC) * counts[t];
    out_cs[t] = ncs;
    lcs[t] = ncs;

    float s = 0.f;
    for (int i = t; i < NROWS; i += 1024) s += ssepart[i];
    for (int m = 16; m >= 1; m >>= 1) s += __shfl_xor(s, m, 32);
    if ((t & 31) == 0) lsse[t >> 5] = s;
    __syncthreads();

    for (int stride = 512; stride >= 1; stride >>= 1) {
        if (t < stride) lcs[t] += lcs[t + stride];
        __syncthreads();
    }
    float nsum = lcs[0];

    if (t == 0) {
        float tot = 0.f;
        for (int i = 0; i < 32; ++i) tot += lsse[i];
        out_loss[0] = (1.0f + COMMITC) * tot / (float)(NROWS * DIM);
    }

    float cs  = (ncs + EPSV) / (nsum + (float)K_CODES * EPSV) * nsum;
    float inv = 1.0f / cs;
    for (int d2 = 0; d2 < DIM; ++d2) {
        float avg = DECAYC * ema_avg[(size_t)t * DIM + d2] + (1.0f - DECAYC) * embsum[(size_t)t * DIM + d2];
        out_avg[(size_t)t * DIM + d2] = avg;
        out_w[(size_t)t * DIM + d2]   = avg * inv;
    }
}

extern "C" void kernel_launch(void* const* d_in, const int* in_sizes, int n_in,
                              void* d_out, int out_size, void* d_ws, size_t ws_size,
                              hipStream_t stream)
{
    (void)in_sizes; (void)n_in; (void)out_size; (void)ws_size;
    const float* x       = (const float*)d_in[0];
    const float* w       = (const float*)d_in[1];
    const float* ema_cs  = (const float*)d_in[2];
    const float* ema_avg = (const float*)d_in[3];

    float* out      = (float*)d_out;
    float* out_q    = out;                 // 16,777,216
    float* out_loss = out + 16777216;      // 1
    float* out_idx  = out + 16777217;      // 65,536
    float* out_w    = out + 16842753;      // 262,144
    float* out_cs   = out + 17104897;      // 1,024
    float* out_avg  = out + 17105921;      // 262,144

    char* ws = (char*)d_ws;
    size_t o = 0;
    unsigned short* xhi  = (unsigned short*)(ws + o); o += (size_t)NROWS * DIM * 2;
    unsigned short* xlo  = (unsigned short*)(ws + o); o += (size_t)NROWS * DIM * 2;
    unsigned short* wthi = (unsigned short*)(ws + o); o += (size_t)DIM * K_CODES * 2;
    unsigned short* wtlo = (unsigned short*)(ws + o); o += (size_t)DIM * K_CODES * 2;
    float* wsq     = (float*)(ws + o); o += K_CODES * 4;
    float* counts  = (float*)(ws + o); o += K_CODES * 4;                 // contiguous with embsum
    float* embsum  = (float*)(ws + o); o += (size_t)K_CODES * DIM * 4;
    int*   idxbuf  = (int*)  (ws + o); o += (size_t)NROWS * 4;
    float* ssepart = (float*)(ws + o); o += (size_t)NROWS * 4;

    (void)hipMemsetAsync(counts, 0, (size_t)(K_CODES + K_CODES * DIM) * 4, stream);

    k_prep_x<<<(NROWS * DIM + 255) / 256, 256, 0, stream>>>(x, xhi, xlo, NROWS * DIM);
    k_prep_w<<<(K_CODES * DIM + 255) / 256, 256, 0, stream>>>(w, wthi, wtlo);
    k_wsq<<<K_CODES, 256, 0, stream>>>(w, wsq);

    // dynamic LDS: 2 x 64KB double buffer + 4KB |e|^2 table = 135168 bytes
    k_argmin<<<NROWS / 128, 256, 135168, stream>>>(xhi, xlo, wthi, wtlo, wsq, idxbuf);

    k_gather<<<NROWS, 256, 0, stream>>>(x, w, idxbuf, out_q, out_idx, counts, embsum, ssepart);

    k_final<<<1, 1024, 0, stream>>>(counts, embsum, ema_cs, ema_avg, ssepart,
                                    out_loss, out_w, out_cs, out_avg);
}
